// WindowAttentionV1_66468913873048
// MI455X (gfx1250) — matrix operand order repaired and checked by reading, not compiled, not run on hardware
//
#include <hip/hip_runtime.h>

typedef __bf16 bf16;
typedef __attribute__((ext_vector_type(16))) __bf16 v16bf;
typedef __attribute__((ext_vector_type(8)))  __bf16 v8bf;
typedef __attribute__((ext_vector_type(8)))  float  v8f;
typedef __attribute__((ext_vector_type(4)))  float  v4f;

#define DIMC 180
#define HEADS 6
#define HD 30
#define HDP 32
#define N64 64
#define NWIN 1024
#define CP 192        // padded channels = HEADS*HDP
#define KT1 6         // 192/32 K-tiles
#define NT1 36        // 576/16 N-tiles (q,k,v packed)
#define NT3 12        // 192/16 N-tiles (proj)
#define SCALE 0.18257418583505536f

// padded LDS row strides (bank-conflict avoidance, 16B alignment preserved)
#define CPS  200      // x/q/k/O rows: 200 bf16 (400 B)
#define VSTR 72       // vT rows: 72 bf16 (144 B)
#define SSTR 68       // S rows: 68 f32 (272 B)
#define PSTR 72       // P rows: 72 bf16 (144 B)

// workspace layout (bytes)
#define WS_WQKV  0
#define WS_WPROJ (WS_WQKV + KT1*NT1*32*16*2)   // 221184
#define WS_BQKV  (WS_WPROJ + KT1*NT3*32*16*2)  // 294912
#define WS_BPROJ (WS_BQKV + 576*4)             // 297216
#define WS_BIASH (WS_BPROJ + 192*4)            // 297984  (+98304 = 396288 total)

// LDS layout (bytes)
#define L_XA 0                     // xa [64][CPS] bf16 = 25600 (reused as O)
#define L_Q  25600                 // q  [64][CPS] bf16
#define L_K  51200                 // k  [64][CPS] bf16
#define L_VT 76800                 // vT [192][VSTR] bf16 = 27648
#define L_S  104448                // S  [64][SSTR] f32 = 17408
#define L_P  121856                // P  [64][PSTR] bf16 = 9216
#define L_END 131072               // 128 KB

__device__ __forceinline__ v16bf cat16(v8bf lo, v8bf hi) {
  return __builtin_shufflevector(lo, hi, 0,1,2,3,4,5,6,7,8,9,10,11,12,13,14,15);
}
__device__ __forceinline__ v8f wmma_bf(v16bf a, v16bf b, v8f c) {
  return __builtin_amdgcn_wmma_f32_16x16x32_bf16(false, a, false, b, (short)0, c, false, false);
}

// ---------------- prep: swizzle weights into WMMA-B layout, fold scale, bias table ----
__global__ void swin_prep_kernel(const float* __restrict__ qkv_w, const float* __restrict__ qkv_b,
                                 const float* __restrict__ proj_w, const float* __restrict__ proj_b,
                                 const float* __restrict__ tbl, const int* __restrict__ rel_idx,
                                 char* __restrict__ ws) {
  bf16*  wqkv  = (bf16*)(ws + WS_WQKV);
  bf16*  wproj = (bf16*)(ws + WS_WPROJ);
  float* bqkv  = (float*)(ws + WS_BQKV);
  float* bproj = (float*)(ws + WS_BPROJ);
  float* biasH = (float*)(ws + WS_BIASH);
  const int NQ = KT1*NT1*512;   // 110592
  const int NP = KT1*NT3*512;   // 36864
  const int NB = HEADS*4096;    // 24576
  const int total = NQ + NP + NB + 576 + 192;
  for (int idx = blockIdx.x*blockDim.x + threadIdx.x; idx < total; idx += gridDim.x*blockDim.x) {
    if (idx < NQ) {
      int e = idx & 15, lane = (idx >> 4) & 31, nt = (idx >> 9) % NT1, kt = idx / (512*NT1);
      int K = kt*32 + ((lane >= 16) ? 8 : 0) + ((e < 8) ? e : 8 + e);
      int n = nt*16 + (lane & 15);
      int which = n / CP, hh = (n % CP) / HDP, d = n & 31;
      float v = 0.f;
      if (d < HD && K < DIMC) v = qkv_w[(which*DIMC + hh*HD + d)*DIMC + K];
      if (which == 0) v *= SCALE;
      wqkv[idx] = (bf16)v;
    } else if (idx < NQ + NP) {
      int j = idx - NQ;
      int e = j & 15, lane = (j >> 4) & 31, nt = (j >> 9) % NT3, kt = j / (512*NT3);
      int kk = kt*32 + ((lane >= 16) ? 8 : 0) + ((e < 8) ? e : 8 + e);      // packed input channel h*32+d
      int h2 = kk >> 5, d2 = kk & 31;
      int o = nt*16 + (lane & 15);
      float v = 0.f;
      if (d2 < HD && o < DIMC) v = proj_w[o*DIMC + h2*HD + d2];
      wproj[j] = (bf16)v;
    } else if (idx < NQ + NP + NB) {
      int j = idx - NQ - NP;
      int h = j >> 12, nm = j & 4095;
      biasH[j] = tbl[rel_idx[nm]*HEADS + h];
    } else if (idx < NQ + NP + NB + 576) {
      int n = idx - NQ - NP - NB;
      int which = n / CP, hh = (n % CP) / HDP, d = n & 31;
      float v = (d < HD) ? qkv_b[which*DIMC + hh*HD + d] : 0.f;
      if (which == 0) v *= SCALE;
      bqkv[n] = v;
    } else {
      int o = idx - NQ - NP - NB - 576;
      bproj[o] = (o < DIMC) ? proj_b[o] : 0.f;
    }
  }
}

// ---------------- fused window attention: one workgroup per window ----------------
__global__ __launch_bounds__(256) void swin_attn_kernel(const float* __restrict__ x,
                                                        const float* __restrict__ mask,
                                                        const char* __restrict__ ws,
                                                        float* __restrict__ out) {
  extern __shared__ char smem[];
  bf16*  xa = (bf16*)(smem + L_XA);
  bf16*  oo = xa;                       // O reuses xa region after GEMM1
  bf16*  qs = (bf16*)(smem + L_Q);
  bf16*  ks = (bf16*)(smem + L_K);
  bf16*  vT = (bf16*)(smem + L_VT);
  float* S  = (float*)(smem + L_S);
  bf16*  P  = (bf16*)(smem + L_P);

  const bf16*  wqkv  = (const bf16*)(ws + WS_WQKV);
  const bf16*  wproj = (const bf16*)(ws + WS_WPROJ);
  const float* bqkv  = (const float*)(ws + WS_BQKV);
  const float* bproj = (const float*)(ws + WS_BPROJ);
  const float* biasH = (const float*)(ws + WS_BIASH);

  const int b    = blockIdx.x;
  const int w    = b & (NWIN - 1);
  const int tid  = threadIdx.x;
  const int lane = tid & 31;
  const int wave = tid >> 5;
  const int l16  = lane & 15;
  const int hi   = (lane >> 4) & 1;

  // ---- phase 0: stage x into LDS as bf16, pad K 180->192 ----
  const float* xb = x + (size_t)b * N64 * DIMC;
  for (int i = tid; i < N64*DIMC; i += 256) {
    int r = i / DIMC, c = i - r*DIMC;
    xa[r*CPS + c] = (bf16)xb[i];
  }
  for (int i = tid; i < N64*(CP-DIMC); i += 256) {
    int r = i / (CP-DIMC), c = DIMC + (i - r*(CP-DIMC));
    xa[r*CPS + c] = (bf16)0.f;
  }
  __syncthreads();

  // ---- phase 1: QKV GEMM  [64x192] x [192x576] -> q,k row-major; v transposed ----
  {
    const int mt = wave & 3;               // uniform per wave
    const int arow = (mt*16 + l16) * CPS + hi*8;
    const int row0 = mt*16 + hi*8;
    for (int nt = wave >> 2; nt < NT1; nt += 2) {
      v8f acc = {};
      #pragma unroll
      for (int kt = 0; kt < KT1; ++kt) {
        v8bf alo = *(const v8bf*)(xa + arow + kt*32);
        v8bf ahi = *(const v8bf*)(xa + arow + kt*32 + 16);
        v16bf bm = *(const v16bf*)(wqkv + ((kt*NT1 + nt)*32 + lane)*16);
        acc = wmma_bf(cat16(alo, ahi), bm, acc);
      }
      float bia = bqkv[nt*16 + l16];
      v8bf pk;
      #pragma unroll
      for (int r = 0; r < 8; ++r) pk[r] = (bf16)(acc[r] + bia);
      // `which` depends only on nt -> force scalar branch (wave-uniform)
      int which = __builtin_amdgcn_readfirstlane(nt / 12);
      int cloc  = nt*16 + l16 - which*CP;
      if (which == 2) {
        *(v8bf*)(vT + cloc*VSTR + row0) = pk;          // packed b128 store
      } else {
        bf16* dst = which ? ks : qs;
        #pragma unroll
        for (int r = 0; r < 8; ++r) dst[(row0 + r)*CPS + cloc] = pk[r];
      }
    }
  }
  __syncthreads();

  // ---- phase 2: per-head attention ----
  const float* maskb = mask + (size_t)w * 4096;
  for (int h = 0; h < HEADS; ++h) {
    const float* biasb = biasH + h*4096;
    // (a) S = q k^T   (4x4 tiles, K=32); nt uniform per wave
    {
      const int nt = wave & 3;
      v8bf blo = *(const v8bf*)(ks + (nt*16 + l16)*CPS + h*HDP + hi*8);
      v8bf bhi = *(const v8bf*)(ks + (nt*16 + l16)*CPS + h*HDP + hi*8 + 16);
      v16bf bm = cat16(blo, bhi);
      for (int mt = wave >> 2; mt < 4; mt += 2) {
        v8f acc = {};
        v8bf alo = *(const v8bf*)(qs + (mt*16 + l16)*CPS + h*HDP + hi*8);
        v8bf ahi = *(const v8bf*)(qs + (mt*16 + l16)*CPS + h*HDP + hi*8 + 16);
        acc = wmma_bf(cat16(alo, ahi), bm, acc);
        int m = nt*16 + l16;
        int row0 = mt*16 + hi*8;
        #pragma unroll
        for (int r = 0; r < 8; ++r) S[(row0 + r)*SSTR + m] = acc[r];
      }
    }
    __syncthreads();
    // (b) softmax rows (4 lanes per row), add rel-pos bias + shift mask from L2
    {
      int row = tid >> 2;
      int cb  = (tid & 3) * 16;
      const v4f* Sr = (const v4f*)(S + row*SSTR + cb);
      const v4f* bi = (const v4f*)(biasb + row*N64 + cb);
      const v4f* mk = (const v4f*)(maskb + row*N64 + cb);
      v4f vv[4];
      float mx = -3.4e38f;
      #pragma unroll
      for (int j = 0; j < 4; ++j) {
        v4f v = Sr[j] + bi[j] + mk[j];
        vv[j] = v;
        mx = fmaxf(mx, fmaxf(fmaxf(v.x, v.y), fmaxf(v.z, v.w)));
      }
      mx = fmaxf(mx, __shfl_xor(mx, 1, 32));
      mx = fmaxf(mx, __shfl_xor(mx, 2, 32));
      float sum = 0.f;
      #pragma unroll
      for (int j = 0; j < 4; ++j) {
        v4f e;
        e.x = __expf(vv[j].x - mx); e.y = __expf(vv[j].y - mx);
        e.z = __expf(vv[j].z - mx); e.w = __expf(vv[j].w - mx);
        vv[j] = e;
        sum += e.x + e.y + e.z + e.w;
      }
      sum += __shfl_xor(sum, 1, 32);
      sum += __shfl_xor(sum, 2, 32);
      float inv = 1.f / sum;
      v8bf pk0, pk1;
      #pragma unroll
      for (int j = 0; j < 4; ++j) {
        v4f p = vv[j] * inv;
        v8bf& pk = (j < 2) ? pk0 : pk1;
        int o = (j & 1) * 4;
        pk[o+0] = (bf16)p.x; pk[o+1] = (bf16)p.y; pk[o+2] = (bf16)p.z; pk[o+3] = (bf16)p.w;
      }
      v8bf* Pr = (v8bf*)(P + row*PSTR + cb);
      Pr[0] = pk0; Pr[1] = pk1;
    }
    __syncthreads();
    // (c) O_h = P V_h  (4x2 tiles, K=64 -> 2 steps); one tile per wave
    {
      int mt = wave >> 1, nt = wave & 1;
      v8f acc = {};
      #pragma unroll
      for (int kt = 0; kt < 2; ++kt) {
        v8bf alo = *(const v8bf*)(P + (mt*16 + l16)*PSTR + kt*32 + hi*8);
        v8bf ahi = *(const v8bf*)(P + (mt*16 + l16)*PSTR + kt*32 + hi*8 + 16);
        v8bf blo = *(const v8bf*)(vT + (h*HDP + nt*16 + l16)*VSTR + kt*32 + hi*8);
        v8bf bhi = *(const v8bf*)(vT + (h*HDP + nt*16 + l16)*VSTR + kt*32 + hi*8 + 16);
        acc = wmma_bf(cat16(alo, ahi), cat16(blo, bhi), acc);
      }
      int row0 = mt*16 + hi*8;
      int col  = h*HDP + nt*16 + l16;
      #pragma unroll
      for (int r = 0; r < 8; ++r) oo[(row0 + r)*CPS + col] = (bf16)acc[r];
    }
    __syncthreads();
  }

  // ---- phase 3: output projection  [64x192] x [192x192] -> out[64x180] ----
  {
    float* outb = out + (size_t)b * N64 * DIMC;
    const int mt = wave & 3;               // uniform per wave
    const int arow = (mt*16 + l16) * CPS + hi*8;
    const int row0 = mt*16 + hi*8;
    for (int nt = wave >> 2; nt < NT3; nt += 2) {
      v8f acc = {};
      #pragma unroll
      for (int kt = 0; kt < KT1; ++kt) {
        v8bf alo = *(const v8bf*)(oo + arow + kt*32);
        v8bf ahi = *(const v8bf*)(oo + arow + kt*32 + 16);
        v16bf bm = *(const v16bf*)(wproj + ((kt*NT3 + nt)*32 + lane)*16);
        acc = wmma_bf(cat16(alo, ahi), bm, acc);
      }
      int col = nt*16 + l16;
      if (col < DIMC) {
        float bia = bproj[col];
        #pragma unroll
        for (int r = 0; r < 8; ++r) outb[(row0 + r)*DIMC + col] = acc[r] + bia;
      }
    }
  }
}

extern "C" void kernel_launch(void* const* d_in, const int* in_sizes, int n_in,
                              void* d_out, int out_size, void* d_ws, size_t ws_size,
                              hipStream_t stream) {
  const float* x      = (const float*)d_in[0];
  const float* mask   = (const float*)d_in[1];
  const float* qkv_w  = (const float*)d_in[2];
  const float* qkv_b  = (const float*)d_in[3];
  const float* proj_w = (const float*)d_in[4];
  const float* proj_b = (const float*)d_in[5];
  const float* tbl    = (const float*)d_in[6];
  const int*   rel    = (const int*)d_in[7];
  char* ws = (char*)d_ws;

  // 675*256 == exact element count of all prep jobs
  swin_prep_kernel<<<675, 256, 0, stream>>>(qkv_w, qkv_b, proj_w, proj_b, tbl, rel, ws);

  int nwin = in_sizes[0] / (N64 * DIMC);   // 8192 windows
  swin_attn_kernel<<<nwin, 256, L_END, stream>>>(x, mask, ws, (float*)d_out);
}
